// Pointnet2Backbone_1434519077265
// MI455X (gfx1250) — compile-verified, matrix operand order checked by reading, not run on hardware
//
#include <hip/hip_runtime.h>
#include <hip/hip_bf16.h>

typedef __attribute__((ext_vector_type(16))) _Float16 v16h;
typedef __attribute__((ext_vector_type(8)))  _Float16 v8h;
typedef __attribute__((ext_vector_type(8)))  float    v8f;

// ---------------------------------------------------------------------------
// Weight prep: f32 W(K,N) -> f16 Wt(N, Kpad) transposed + zero-padded.
// ---------------------------------------------------------------------------
__global__ void transpose_pad_f16(const float* __restrict__ W, _Float16* __restrict__ Wt,
                                  int K, int N, int Kpad) {
  int i = blockIdx.x * blockDim.x + threadIdx.x;
  if (i >= N * Kpad) return;
  int n = i / Kpad, k = i % Kpad;
  Wt[i] = (k < K) ? (_Float16)W[(size_t)k * N + n] : (_Float16)0.0f;
}

// ---------------------------------------------------------------------------
// Farthest point sampling: one block per batch, points+dist in VGPRs,
// wave32 shfl reduce + cross-wave LDS reduce per serial iteration.
// ---------------------------------------------------------------------------
#define FPS_BD 1024
#define FPS_MAXP 20   // ceil(20000/1024)

__global__ __launch_bounds__(FPS_BD)
void fps_kernel(const float* __restrict__ xyz, int B, int N, int npoint, int* __restrict__ inds) {
  int b = blockIdx.x;
  int tid = threadIdx.x;
  const float* base = xyz + (size_t)b * N * 3;

  float px[FPS_MAXP], py[FPS_MAXP], pz[FPS_MAXP], dist[FPS_MAXP];
#pragma unroll
  for (int i = 0; i < FPS_MAXP; i++) {
    int p = i * FPS_BD + tid;
    if (p < N) {
      px[i] = base[p * 3 + 0]; py[i] = base[p * 3 + 1]; pz[i] = base[p * 3 + 2];
      dist[i] = 1e10f;
    } else {
      px[i] = 0.f; py[i] = 0.f; pz[i] = 0.f; dist[i] = -1.0f;
    }
  }

  __shared__ float s_val[FPS_BD / 32];
  __shared__ int   s_idx[FPS_BD / 32];
  __shared__ int   s_best;
  __shared__ float s_seed[3];

  if (tid == 0) {
    inds[(size_t)b * npoint] = 0;
    s_seed[0] = base[0]; s_seed[1] = base[1]; s_seed[2] = base[2];
  }
  __syncthreads();

  for (int it = 1; it < npoint; it++) {
    float sx = s_seed[0], sy = s_seed[1], sz = s_seed[2];
    float bv = -1e30f; int bi = 0;
#pragma unroll
    for (int i = 0; i < FPS_MAXP; i++) {
      int p = i * FPS_BD + tid;
      if (p < N) {
        float dx = px[i] - sx, dy = py[i] - sy, dz = pz[i] - sz;
        float d = dx * dx + dy * dy + dz * dz;
        if (d < dist[i]) dist[i] = d;
        if (dist[i] > bv) { bv = dist[i]; bi = p; }
      }
    }
    // wave32 reduce (argmax, ties -> lower index like jnp.argmax)
#pragma unroll
    for (int o = 16; o > 0; o >>= 1) {
      float ov = __shfl_down(bv, o, 32);
      int   oi = __shfl_down(bi, o, 32);
      if (ov > bv || (ov == bv && oi < bi)) { bv = ov; bi = oi; }
    }
    __syncthreads();
    if ((tid & 31) == 0) { s_val[tid >> 5] = bv; s_idx[tid >> 5] = bi; }
    __syncthreads();
    if (tid < 32) {
      bv = s_val[tid]; bi = s_idx[tid];
#pragma unroll
      for (int o = 16; o > 0; o >>= 1) {
        float ov = __shfl_down(bv, o, 32);
        int   oi = __shfl_down(bi, o, 32);
        if (ov > bv || (ov == bv && oi < bi)) { bv = ov; bi = oi; }
      }
      if (tid == 0) {
        s_best = bi;
        inds[(size_t)b * npoint + it] = bi;
        s_seed[0] = base[bi * 3 + 0];
        s_seed[1] = base[bi * 3 + 1];
        s_seed[2] = base[bi * 3 + 2];
      }
    }
    __syncthreads();
  }
}

// ---------------------------------------------------------------------------
// Gather sampled centers.
// ---------------------------------------------------------------------------
__global__ void gather_xyz_kernel(const float* __restrict__ xyz, const int* __restrict__ inds,
                                  int B, int N, int S, float* __restrict__ out) {
  int i = blockIdx.x * blockDim.x + threadIdx.x;
  if (i >= B * S * 3) return;
  int k = i % 3, g = i / 3, b = g / S;
  out[i] = xyz[((size_t)b * N + inds[g]) * 3 + k];
}

// ---------------------------------------------------------------------------
// Ball query: one wave32 per query point; indices emitted in ascending order
// via ballot/popc rank compaction (matches reference top_k(-index) semantics).
// ---------------------------------------------------------------------------
__global__ void ball_query_kernel(const float* __restrict__ xyz, const float* __restrict__ new_xyz,
                                  int B, int N, int S, float r2, int nsample,
                                  int* __restrict__ idx) {
  int wid = (blockIdx.x * blockDim.x + threadIdx.x) >> 5;
  int lane = threadIdx.x & 31;
  if (wid >= B * S) return;
  int b = wid / S;
  const float* q = new_xyz + (size_t)wid * 3;
  float qx = q[0], qy = q[1], qz = q[2];
  const float* px = xyz + (size_t)b * N * 3;
  int* out = idx + (size_t)wid * nsample;

  int found = 0, first = 0;
  for (int base = 0; base < N && found < nsample; base += 32) {
    int p = base + lane;
    bool valid = false;
    if (p < N) {
      float dx = px[p * 3 + 0] - qx, dy = px[p * 3 + 1] - qy, dz = px[p * 3 + 2] - qz;
      valid = (dx * dx + dy * dy + dz * dz) < r2;
    }
    unsigned mask = (unsigned)__ballot(valid);
    if (found == 0 && mask) first = base + (__ffs(mask) - 1);
    int rank = __popc(mask & ((1u << lane) - 1u));
    if (valid && (found + rank) < nsample) out[found + rank] = p;
    found += __popc(mask);
  }
  if (found > nsample) found = nsample;
  if (lane == 0)
    for (int j = found; j < nsample; j++) out[j] = first;
}

// ---------------------------------------------------------------------------
// Build grouped activation matrix G(f16): row = (b,s,j), cols = [(xyz-c)/r | feats | 0-pad]
// ---------------------------------------------------------------------------
__global__ void group_kernel(const float* __restrict__ xyz, const float* __restrict__ new_xyz,
                             const int* __restrict__ idx, const float* __restrict__ feats,
                             int B, int N, int S, int ns, int C, int Kpad, float inv_r,
                             _Float16* __restrict__ G) {
  size_t total = (size_t)B * S * ns * Kpad;
  size_t i = (size_t)blockIdx.x * blockDim.x + threadIdx.x;
  if (i >= total) return;
  int k = (int)(i % Kpad);
  size_t row = i / Kpad;
  size_t g = row / ns;          // b*S + s
  int b = (int)(g / S);
  int pi = idx[row];
  float v = 0.0f;
  if (k < 3) {
    v = (xyz[((size_t)b * N + pi) * 3 + k] - new_xyz[g * 3 + k]) * inv_r;
  } else if (k - 3 < C) {
    v = feats[((size_t)b * N + pi) * C + (k - 3)];
  }
  G[i] = (_Float16)v;
}

// ---------------------------------------------------------------------------
// WMMA GEMM: Y = relu(A(M,K) * Wt(N,K)^T + bias). One wave32 per 16x64 strip
// (4 N-tiles): A fragment loaded once per k-step, reused across 4 WMMAs ->
// 2.5 B128 loads per v_wmma instead of 4. K multiple of 32; f32 accumulate.
// Fragment layout per CDNA5 ISA 7.12.2: lane = {half=lane>>4, m/n=lane&15},
// elems 0..7 -> K = half*8+e, elems 8..15 -> K = 16+half*8+e.
// ---------------------------------------------------------------------------
__global__ __launch_bounds__(32)
void gemm_bias_relu_wmma(const _Float16* __restrict__ A, const _Float16* __restrict__ Wt,
                         const float* __restrict__ bias,
                         _Float16* __restrict__ out16, float* __restrict__ out32,
                         int M, int N, int K) {
  int tm = blockIdx.x;
  int tn0 = blockIdx.y * 4;          // first of 4 N-tiles
  int lane = threadIdx.x;
  int half = lane >> 4;
  int r = lane & 15;
  const _Float16* arow = A + (size_t)(tm * 16 + r) * K;
  const _Float16* brow[4];
#pragma unroll
  for (int j = 0; j < 4; j++)
    brow[j] = Wt + (size_t)((tn0 + j) * 16 + r) * K;

  v8f acc[4] = {};
  for (int k0 = 0; k0 < K; k0 += 32) {
    if (k0 + 128 < K) {
      __builtin_prefetch(arow + k0 + 128, 0, 1);
      __builtin_prefetch(brow[0] + k0 + 128, 0, 1);
    }
    v8h a_lo = *(const v8h*)(arow + k0 + half * 8);
    v8h a_hi = *(const v8h*)(arow + k0 + 16 + half * 8);
    v16h a;
#pragma unroll
    for (int e = 0; e < 8; e++) { a[e] = a_lo[e]; a[e + 8] = a_hi[e]; }
#pragma unroll
    for (int j = 0; j < 4; j++) {
      v8h b_lo = *(const v8h*)(brow[j] + k0 + half * 8);
      v8h b_hi = *(const v8h*)(brow[j] + k0 + 16 + half * 8);
      v16h bb;
#pragma unroll
      for (int e = 0; e < 8; e++) { bb[e] = b_lo[e]; bb[e + 8] = b_hi[e]; }
      acc[j] = __builtin_amdgcn_wmma_f32_16x16x32_f16(false, a, false, bb,
                                                      (short)0, acc[j], false, false);
    }
  }

  int m0 = tm * 16 + half * 8;
#pragma unroll
  for (int j = 0; j < 4; j++) {
    int n = (tn0 + j) * 16 + r;
    float bval = bias ? bias[n] : 0.0f;
#pragma unroll
    for (int v = 0; v < 8; v++) {
      float y = acc[j][v] + bval;
      y = y > 0.0f ? y : 0.0f;
      size_t o = (size_t)(m0 + v) * N + n;
      if (out16) out16[o] = (_Float16)y;
      if (out32) out32[o] = y;
    }
  }
}

// ---------------------------------------------------------------------------
// Max pool over nsample -> f32 pooled features.
// ---------------------------------------------------------------------------
__global__ void maxpool_kernel(const _Float16* __restrict__ G, int groups, int ns, int C,
                               float* __restrict__ out) {
  int i = blockIdx.x * blockDim.x + threadIdx.x;
  if (i >= groups * C) return;
  int g = i / C, c = i % C;
  const _Float16* p = G + (size_t)g * ns * C + c;
  float m = -1e30f;
  for (int j = 0; j < ns; j++) {
    float v = (float)p[(size_t)j * C];
    m = v > m ? v : m;
  }
  out[i] = m;
}

// ---------------------------------------------------------------------------
// 3-NN search (thread per upsampled point, serial scan of <=512 candidates).
// ---------------------------------------------------------------------------
__global__ void three_nn_kernel(const float* __restrict__ xyz1, const float* __restrict__ xyz2,
                                int B, int n1, int n2,
                                int* __restrict__ nn_idx, float* __restrict__ nn_w) {
  int i = blockIdx.x * blockDim.x + threadIdx.x;
  if (i >= B * n1) return;
  int b = i / n1;
  const float* p = xyz1 + (size_t)i * 3;
  float qx = p[0], qy = p[1], qz = p[2];
  const float* q = xyz2 + (size_t)b * n2 * 3;
  float d0 = 1e30f, d1 = 1e30f, d2 = 1e30f;
  int i0 = 0, i1 = 0, i2 = 0;
  for (int j = 0; j < n2; j++) {
    float dx = q[j * 3 + 0] - qx, dy = q[j * 3 + 1] - qy, dz = q[j * 3 + 2] - qz;
    float d = dx * dx + dy * dy + dz * dz;
    if (d < d0)      { d2 = d1; i2 = i1; d1 = d0; i1 = i0; d0 = d; i0 = j; }
    else if (d < d1) { d2 = d1; i2 = i1; d1 = d;  i1 = j; }
    else if (d < d2) { d2 = d;  i2 = j; }
  }
  float w0 = 1.0f / (d0 + 1e-8f), w1 = 1.0f / (d1 + 1e-8f), w2 = 1.0f / (d2 + 1e-8f);
  float ws = w0 + w1 + w2;
  nn_idx[i * 3 + 0] = i0; nn_idx[i * 3 + 1] = i1; nn_idx[i * 3 + 2] = i2;
  nn_w[i * 3 + 0] = w0 / ws; nn_w[i * 3 + 1] = w1 / ws; nn_w[i * 3 + 2] = w2 / ws;
}

// ---------------------------------------------------------------------------
// Interpolate feats2 with 3-NN weights, concat with feats1 -> G f16 (cols: interp|feats1).
// ---------------------------------------------------------------------------
__global__ void interp_concat_kernel(const float* __restrict__ feats1, const float* __restrict__ feats2,
                                     const int* __restrict__ nn_idx, const float* __restrict__ nn_w,
                                     int B, int n1, int n2, int C1, int C2,
                                     _Float16* __restrict__ G) {
  int C = C1 + C2;
  int i = blockIdx.x * blockDim.x + threadIdx.x;
  if (i >= B * n1 * C) return;
  int r = i / C, c = i % C;
  int b = r / n1;
  float v;
  if (c < C2) {
    const int* id = nn_idx + r * 3;
    const float* w = nn_w + r * 3;
    const float* f2 = feats2 + (size_t)b * n2 * C2;
    v = w[0] * f2[(size_t)id[0] * C2 + c] +
        w[1] * f2[(size_t)id[1] * C2 + c] +
        w[2] * f2[(size_t)id[2] * C2 + c];
  } else {
    v = feats1[(size_t)r * C1 + (c - C2)];
  }
  G[(size_t)r * C + c] = (_Float16)v;
}

// ---------------------------------------------------------------------------
// Host orchestration
// ---------------------------------------------------------------------------
static inline size_t align256(size_t x) { return (x + 255) & ~(size_t)255; }
static inline int cdiv(int a, int b) { return (a + b - 1) / b; }

extern "C" void kernel_launch(void* const* d_in, const int* in_sizes, int n_in,
                              void* d_out, int out_size, void* d_ws, size_t ws_size,
                              hipStream_t stream) {
  (void)in_sizes; (void)n_in; (void)out_size; (void)ws_size;
  const int B = 2, N0 = 20000;
  const float* pc = (const float*)d_in[0];

  // 16 MLP layers: {weight d_in index, K, N}; bias = d_in[w+1]
  struct LD { int w, K, N; };
  const LD L[16] = {
    {1, 3, 64},   {3, 64, 64},   {5, 64, 128},     // sa1
    {7, 131, 128},{9, 128, 128}, {11, 128, 256},   // sa2
    {13, 259, 128},{15, 128, 128},{17, 128, 256},  // sa3
    {19, 259, 128},{21, 128, 128},{23, 128, 256},  // sa4
    {25, 512, 256},{27, 256, 256},                 // fp1
    {29, 512, 256},{31, 256, 256}                  // fp2
  };
  int Kp[16];
  for (int l = 0; l < 16; l++) Kp[l] = ((L[l].K + 31) / 32) * 32;

  // ---- workspace carve ----
  char* ws = (char*)d_ws;
  size_t off = 0;
  auto carve = [&](size_t bytes) -> void* {
    void* p = ws + off; off = align256(off + bytes); return p;
  };
  _Float16* wt[16];
  for (int l = 0; l < 16; l++) wt[l] = (_Float16*)carve((size_t)L[l].N * Kp[l] * 2);
  _Float16* Ga = (_Float16*)carve((size_t)16777216 * 2);  // 32 MB (max M*64)
  _Float16* Gb = (_Float16*)carve((size_t)33554432 * 2);  // 64 MB (max M*128)
  float* nx1 = (float*)carve((size_t)B * 2048 * 3 * 4);
  float* nx2 = (float*)carve((size_t)B * 1024 * 3 * 4);
  float* nx3 = (float*)carve((size_t)B * 512 * 3 * 4);
  float* nx4 = (float*)carve((size_t)B * 256 * 3 * 4);
  int* inds1 = (int*)carve((size_t)B * 2048 * 4);
  int* inds2 = (int*)carve((size_t)B * 1024 * 4);
  int* inds3 = (int*)carve((size_t)B * 512 * 4);
  int* inds4 = (int*)carve((size_t)B * 256 * 4);
  int* idx1 = (int*)carve((size_t)B * 2048 * 64 * 4);
  int* idx2 = (int*)carve((size_t)B * 1024 * 32 * 4);
  int* idx3 = (int*)carve((size_t)B * 512 * 16 * 4);
  int* idx4 = (int*)carve((size_t)B * 256 * 16 * 4);
  float* f1 = (float*)carve((size_t)B * 2048 * 128 * 4);
  float* f2 = (float*)carve((size_t)B * 1024 * 256 * 4);
  float* f3 = (float*)carve((size_t)B * 512 * 256 * 4);
  float* f4 = (float*)carve((size_t)B * 256 * 256 * 4);
  float* fp1o = (float*)carve((size_t)B * 512 * 256 * 4);
  int*   nni = (int*)carve((size_t)B * 1024 * 3 * 4);
  float* nnw = (float*)carve((size_t)B * 1024 * 3 * 4);

  // ---- weight prep ----
  for (int l = 0; l < 16; l++) {
    int tot = L[l].N * Kp[l];
    transpose_pad_f16<<<cdiv(tot, 256), 256, 0, stream>>>(
        (const float*)d_in[L[l].w], wt[l], L[l].K, L[l].N, Kp[l]);
  }

  auto gemm = [&](const _Float16* A, int l, _Float16* o16, float* o32, int M) {
    dim3 grid(M / 16, L[l].N / 64);   // each wave: 16x64 strip (4 N-tiles)
    gemm_bias_relu_wmma<<<grid, 32, 0, stream>>>(
        A, wt[l], (const float*)d_in[L[l].w + 1], o16, o32, M, L[l].N, Kp[l]);
  };

  // ================= SA1: N=20000 -> S=2048, r=0.04, ns=64, [3->64->64->128]
  {
    const int N = N0, S = 2048, ns = 64; const float r = 0.04f;
    fps_kernel<<<B, FPS_BD, 0, stream>>>(pc, B, N, S, inds1);
    gather_xyz_kernel<<<cdiv(B * S * 3, 256), 256, 0, stream>>>(pc, inds1, B, N, S, nx1);
    ball_query_kernel<<<cdiv(B * S * 32, 256), 256, 0, stream>>>(pc, nx1, B, N, S, r * r, ns, idx1);
    int M = B * S * ns;
    group_kernel<<<cdiv(M * 32, 256), 256, 0, stream>>>(pc, nx1, idx1, nullptr, B, N, S, ns, 0, 32, 1.0f / r, Ga);
    gemm(Ga, 0, Gb, nullptr, M);
    gemm(Gb, 1, Ga, nullptr, M);
    gemm(Ga, 2, Gb, nullptr, M);
    maxpool_kernel<<<cdiv(B * S * 128, 256), 256, 0, stream>>>(Gb, B * S, ns, 128, f1);
  }
  // ================= SA2: 2048 -> 1024, r=0.1, ns=32, feats C=128, [131->128->128->256]
  {
    const int N = 2048, S = 1024, ns = 32, C = 128, KP = 160; const float r = 0.1f;
    fps_kernel<<<B, FPS_BD, 0, stream>>>(nx1, B, N, S, inds2);
    gather_xyz_kernel<<<cdiv(B * S * 3, 256), 256, 0, stream>>>(nx1, inds2, B, N, S, nx2);
    ball_query_kernel<<<cdiv(B * S * 32, 256), 256, 0, stream>>>(nx1, nx2, B, N, S, r * r, ns, idx2);
    int M = B * S * ns;
    group_kernel<<<cdiv(M * KP, 256), 256, 0, stream>>>(nx1, nx2, idx2, f1, B, N, S, ns, C, KP, 1.0f / r, Ga);
    gemm(Ga, 3, Gb, nullptr, M);
    gemm(Gb, 4, Ga, nullptr, M);
    gemm(Ga, 5, Gb, nullptr, M);
    maxpool_kernel<<<cdiv(B * S * 256, 256), 256, 0, stream>>>(Gb, B * S, ns, 256, f2);
  }
  // ================= SA3: 1024 -> 512, r=0.2, ns=16, C=256, [259->128->128->256]
  {
    const int N = 1024, S = 512, ns = 16, C = 256, KP = 288; const float r = 0.2f;
    fps_kernel<<<B, FPS_BD, 0, stream>>>(nx2, B, N, S, inds3);
    gather_xyz_kernel<<<cdiv(B * S * 3, 256), 256, 0, stream>>>(nx2, inds3, B, N, S, nx3);
    ball_query_kernel<<<cdiv(B * S * 32, 256), 256, 0, stream>>>(nx2, nx3, B, N, S, r * r, ns, idx3);
    int M = B * S * ns;
    group_kernel<<<cdiv(M * KP, 256), 256, 0, stream>>>(nx2, nx3, idx3, f2, B, N, S, ns, C, KP, 1.0f / r, Ga);
    gemm(Ga, 6, Gb, nullptr, M);
    gemm(Gb, 7, Ga, nullptr, M);
    gemm(Ga, 8, Gb, nullptr, M);
    maxpool_kernel<<<cdiv(B * S * 256, 256), 256, 0, stream>>>(Gb, B * S, ns, 256, f3);
  }
  // ================= SA4: 512 -> 256, r=0.3, ns=16, C=256, [259->128->128->256]
  {
    const int N = 512, S = 256, ns = 16, C = 256, KP = 288; const float r = 0.3f;
    fps_kernel<<<B, FPS_BD, 0, stream>>>(nx3, B, N, S, inds4);
    gather_xyz_kernel<<<cdiv(B * S * 3, 256), 256, 0, stream>>>(nx3, inds4, B, N, S, nx4);
    ball_query_kernel<<<cdiv(B * S * 32, 256), 256, 0, stream>>>(nx3, nx4, B, N, S, r * r, ns, idx4);
    int M = B * S * ns;
    group_kernel<<<cdiv(M * KP, 256), 256, 0, stream>>>(nx3, nx4, idx4, f3, B, N, S, ns, C, KP, 1.0f / r, Ga);
    gemm(Ga, 9, Gb, nullptr, M);
    gemm(Gb, 10, Ga, nullptr, M);
    gemm(Ga, 11, Gb, nullptr, M);
    maxpool_kernel<<<cdiv(B * S * 256, 256), 256, 0, stream>>>(Gb, B * S, ns, 256, f4);
  }
  // ================= FP1: (xyz3, xyz4, f3, f4) [512->256->256] -> fp1o
  {
    const int n1 = 512, n2 = 256, C1 = 256, C2 = 256;
    three_nn_kernel<<<cdiv(B * n1, 256), 256, 0, stream>>>(nx3, nx4, B, n1, n2, nni, nnw);
    interp_concat_kernel<<<cdiv(B * n1 * (C1 + C2), 256), 256, 0, stream>>>(
        f3, f4, nni, nnw, B, n1, n2, C1, C2, Ga);
    int M = B * n1;
    gemm(Ga, 12, Gb, nullptr, M);
    gemm(Gb, 13, nullptr, fp1o, M);
  }
  // ================= FP2: (xyz2, xyz3, f2, fp1o) [512->256->256] -> d_out
  {
    const int n1 = 1024, n2 = 512, C1 = 256, C2 = 256;
    three_nn_kernel<<<cdiv(B * n1, 256), 256, 0, stream>>>(nx2, nx3, B, n1, n2, nni, nnw);
    interp_concat_kernel<<<cdiv(B * n1 * (C1 + C2), 256), 256, 0, stream>>>(
        f2, fp1o, nni, nnw, B, n1, n2, C1, C2, Ga);
    int M = B * n1;
    gemm(Ga, 14, Gb, nullptr, M);
    gemm(Gb, 15, nullptr, (float*)d_out, M);
  }
}
